// RingDilatedAttention_8615704396434
// MI455X (gfx1250) — matrix operand order repaired and checked by reading, not compiled, not run on hardware
//
#include <hip/hip_runtime.h>
#include <hip/hip_bf16.h>

typedef __attribute__((ext_vector_type(16))) __bf16 v16bf;
typedef __attribute__((ext_vector_type(8)))  float   v8f;
typedef __attribute__((ext_vector_type(8)))  __bf16  bf16x8;
typedef __attribute__((ext_vector_type(4)))  __bf16  bf16x4;

#define N_SEQ  8192
#define N_HEAD 12
#define HDIM   64
#define KEYS   2048   // dilated keys per segment (s/r == 2048 for all groups)
#define KCHUNK 64     // keys staged per phase
#define KPER   128    // keys per softmax period (2 phases)
#define QBLK   128    // queries per workgroup
#define WAVES  8

__device__ __forceinline__ float rmax16(float v){
  v = fmaxf(v, __shfl_xor(v, 1, 32));
  v = fmaxf(v, __shfl_xor(v, 2, 32));
  v = fmaxf(v, __shfl_xor(v, 4, 32));
  v = fmaxf(v, __shfl_xor(v, 8, 32));
  return v;
}
__device__ __forceinline__ bf16x8 cvt8(float4 a, float4 b){
  bf16x8 r;
  r[0]=(__bf16)a.x; r[1]=(__bf16)a.y; r[2]=(__bf16)a.z; r[3]=(__bf16)a.w;
  r[4]=(__bf16)b.x; r[5]=(__bf16)b.y; r[6]=(__bf16)b.z; r[7]=(__bf16)b.w;
  return r;
}
__device__ __forceinline__ v16bf cat16(bf16x8 lo, bf16x8 hi){
  v16bf r;
#pragma unroll
  for(int i=0;i<8;i++){ r[i]=lo[i]; r[i+8]=hi[i]; }
  return r;
}

__global__ __launch_bounds__(256)
void RingDilatedAttention_8615704396434_kernel(const float* __restrict__ q,
                                               const float* __restrict__ k,
                                               const float* __restrict__ v,
                                               float* __restrict__ out)
{
  __shared__ __align__(16) __bf16 Ks[KCHUNK*HDIM];        // [key][d], re-staged per phase
  __shared__ __align__(16) __bf16 Vt[2][HDIM*KCHUNK];     // [d][key], double-buffered
  __shared__ __align__(16) __bf16 Pscr[WAVES][16*KPER];   // wave-private P (16 x 128)
  __shared__ float Mscr[WAVES][16];                       // row->lane alpha transpose

  const int tid  = threadIdx.x;
  const int lane = tid & 31;
  const int wave = tid >> 5;
  const int ln   = lane & 15;   // col lane index
  const int hh   = lane >> 4;   // half-wave selector

  // block decode: grp(3) x batch(2) x head_in_group(4) x qblock(64)
  const int blk  = blockIdx.x;
  const int grp  = blk >> 9;
  const int rem  = blk & 511;
  const int bat  = rem >> 8;
  const int rem2 = rem & 255;
  const int head = grp*4 + (rem2 >> 6);
  const int qb   = rem2 & 63;

  const int s   = 2048 << grp;  // segment length
  const int r   = 1    << grp;  // dilation
  const int off = grp;          // i % r == i for (1,2,4)
  const int q0  = qb * QBLK;
  const int segbase = q0 & ~(s - 1);

  // softmax scale folded into Q: scores emerge in exp2 domain
  const float qscale = 0.125f * 1.44269504088896340736f; // 1/sqrt(64) * log2(e)

  // ---- Q tile (16 queries per wave), pre-scaled, in A-fragment layout ----
  const int qrow = q0 + wave*16 + ln;
  const float* qp = q + ((bat*N_SEQ + qrow)*N_HEAD + head)*HDIM;
  v16bf aQ[2];
#pragma unroll
  for(int c=0;c<2;c++){
    const float* p0 = qp + 32*c + 8*hh;
    float4 a0 = *(const float4*)(p0);
    float4 a1 = *(const float4*)(p0 + 4);
    float4 b0 = *(const float4*)(p0 + 16);
    float4 b1 = *(const float4*)(p0 + 20);
    a0.x*=qscale; a0.y*=qscale; a0.z*=qscale; a0.w*=qscale;
    a1.x*=qscale; a1.y*=qscale; a1.z*=qscale; a1.w*=qscale;
    b0.x*=qscale; b0.y*=qscale; b0.z*=qscale; b0.w*=qscale;
    b1.x*=qscale; b1.y*=qscale; b1.z*=qscale; b1.w*=qscale;
    aQ[c] = cat16(cvt8(a0,a1), cvt8(b0,b1));
  }

  // all-ones A fragment: L = ones(16x32) x P^T broadcasts row-sum over all rows
  v16bf aOnes;
#pragma unroll
  for(int i=0;i<16;i++) aOnes[i] = (__bf16)1.0f;

  v8f   O[4];                 // O^T: 4 blocks of (16 d-rows x 16 query-cols)
  v8f   Lacc = {};            // per-query l, broadcast across rows
  float m_[8];                // running max, row domain (row j+8*hh)
#pragma unroll
  for(int d=0;d<4;d++){ v8f z = {}; O[d] = z; }
#pragma unroll
  for(int j=0;j<8;j++) m_[j] = -1e30f;

  // loader work split
  const int kkey = tid >> 2;            // 0..63
  const int kd   = (tid & 3) * 16;      // 0,16,32,48
  const int vkq  = (tid & 15) * 4;      // key quad base
  const int vdq  = (tid >> 4) * 4;      // d quad base

  // ---- software pipeline: global loads for a 64-key phase staged in registers ----
  float4 kf0,kf1,kf2,kf3, vf0,vf1,vf2,vf3;
  auto issue_loads = [&](int kc2){
    const int seqk = segbase + off + (kc2 + kkey)*r;
    const float* kp = k + ((bat*N_SEQ + seqk)*N_HEAD + head)*HDIM + kd;
    kf0 = *(const float4*)(kp);
    kf1 = *(const float4*)(kp + 4);
    kf2 = *(const float4*)(kp + 8);
    kf3 = *(const float4*)(kp + 12);
    const int sv0 = segbase + off + (kc2 + vkq    )*r;
    const int sv1 = segbase + off + (kc2 + vkq + 1)*r;
    const int sv2 = segbase + off + (kc2 + vkq + 2)*r;
    const int sv3 = segbase + off + (kc2 + vkq + 3)*r;
    vf0 = *(const float4*)(v + ((bat*N_SEQ + sv0)*N_HEAD + head)*HDIM + vdq);
    vf1 = *(const float4*)(v + ((bat*N_SEQ + sv1)*N_HEAD + head)*HDIM + vdq);
    vf2 = *(const float4*)(v + ((bat*N_SEQ + sv2)*N_HEAD + head)*HDIM + vdq);
    vf3 = *(const float4*)(v + ((bat*N_SEQ + sv3)*N_HEAD + head)*HDIM + vdq);
    if (kc2 + KCHUNK < KEYS) {  // warm L2 one phase ahead
      const int seqk3 = segbase + off + (kc2 + KCHUNK + kkey)*r;
      __builtin_prefetch(k + ((bat*N_SEQ + seqk3)*N_HEAD + head)*HDIM + kd, 0, 3);
      const int seqv3 = segbase + off + (kc2 + KCHUNK + vkq)*r;
      __builtin_prefetch(v + ((bat*N_SEQ + seqv3)*N_HEAD + head)*HDIM + vdq, 0, 3);
    }
  };

  auto stage_to_lds = [&](int buf){
    *(bf16x8*)&Ks[kkey*HDIM + kd]     = cvt8(kf0,kf1);
    *(bf16x8*)&Ks[kkey*HDIM + kd + 8] = cvt8(kf2,kf3);
    float vb[4][4];
    vb[0][0]=vf0.x; vb[0][1]=vf0.y; vb[0][2]=vf0.z; vb[0][3]=vf0.w;
    vb[1][0]=vf1.x; vb[1][1]=vf1.y; vb[1][2]=vf1.z; vb[1][3]=vf1.w;
    vb[2][0]=vf2.x; vb[2][1]=vf2.y; vb[2][2]=vf2.z; vb[2][3]=vf2.w;
    vb[3][0]=vf3.x; vb[3][1]=vf3.y; vb[3][2]=vf3.z; vb[3][3]=vf3.w;
#pragma unroll
    for(int dd=0; dd<4; dd++){
      bf16x4 pk;
      pk[0]=(__bf16)vb[0][dd]; pk[1]=(__bf16)vb[1][dd];
      pk[2]=(__bf16)vb[2][dd]; pk[3]=(__bf16)vb[3][dd];
      *(bf16x4*)&Vt[buf][(vdq+dd)*KCHUNK + vkq] = pk;
    }
  };

  auto compute_S = [&](v8f* tt){
#pragma unroll
    for(int sub=0; sub<4; sub++){
      v8f acc = {};
#pragma unroll
      for(int c=0;c<2;c++){
        const __bf16* kr = &Ks[(16*sub + ln)*HDIM + 32*c + 8*hh];
        v16bf bk = cat16(*(const bf16x8*)kr, *(const bf16x8*)(kr + 16));
        acc = __builtin_amdgcn_wmma_f32_16x16x32_bf16(false, aQ[c], false, bk,
                                                      (short)0, acc, false, false);
      }
      tt[sub] = acc;
    }
  };

  issue_loads(0);

#pragma unroll 1
  for(int kc = 0; kc < KEYS; kc += KPER){
    v8f t[8];
    // ---- phase A: keys [kc, kc+64) ----
    stage_to_lds(0);
    __syncthreads();                    // Ks + Vt[0] ready
    issue_loads(kc + KCHUNK);
    compute_S(&t[0]);
    __syncthreads();                    // all waves done reading Ks
    // ---- phase B: keys [kc+64, kc+128) ----
    stage_to_lds(1);
    __syncthreads();                    // Ks + Vt[1] ready
    if (kc + KPER < KEYS) issue_loads(kc + KPER);
    compute_S(&t[4]);

    // ---- one softmax update per 128 keys (row domain) ----
    __bf16* ps = &Pscr[wave][0];
    float*  ms = &Mscr[wave][0];
#pragma unroll
    for(int j=0;j<8;j++){
      float mc = fmaxf(fmaxf(fmaxf(t[0][j], t[1][j]), fmaxf(t[2][j], t[3][j])),
                       fmaxf(fmaxf(t[4][j], t[5][j]), fmaxf(t[6][j], t[7][j])));
      mc = rmax16(mc);
      const float mn = fmaxf(m_[j], mc);
      if (ln == 0) ms[j + 8*hh] = m_[j] - mn;   // row->lane alpha transpose
      m_[j] = mn;
#pragma unroll
      for(int sub=0; sub<8; sub++)
        ps[(j + 8*hh)*KPER + 16*sub + ln] = (__bf16)exp2f(t[sub][j] - mn);
    }

    // alpha per query column (= lane), single exp2; uniform vector rescale
    const float alpha = exp2f(ms[ln]);
    Lacc = Lacc * alpha;
#pragma unroll
    for(int db=0; db<4; db++) O[db] = O[db] * alpha;

    // ---- O^T += V_tile x P^T over four 32-key halves (+ row-sum WMMA) ----
#pragma unroll
    for(int half=0; half<4; half++){
      const int kb  = half*32;            // within the 128-key P tile
      const int kb2 = (half & 1)*32;      // within the 64-key V buffer
      const __bf16* pb = ps + ln*KPER + kb + 16*hh;     // B: P^T (keys x queries)
      v16bf bP = cat16(*(const bf16x8*)pb, *(const bf16x8*)(pb + 8));
      Lacc = __builtin_amdgcn_wmma_f32_16x16x32_bf16(false, aOnes, false, bP,
                                                     (short)0, Lacc, false, false);
      const __bf16* vtb = &Vt[half >> 1][0];
#pragma unroll
      for(int db=0; db<4; db++){
        const __bf16* vr = vtb + (16*db + ln)*KCHUNK + kb2 + 16*hh;  // A: V tile
        v16bf aV = cat16(*(const bf16x8*)vr, *(const bf16x8*)(vr + 8));
        O[db] = __builtin_amdgcn_wmma_f32_16x16x32_bf16(false, aV, false, bP,
                                                        (short)0, O[db], false, false);
      }
    }
    __syncthreads();                    // all waves done reading Ks/Vt this period
  }

  // ---- normalize (one divide per lane) and write out vectorized ----
  const float inv = 1.0f / Lacc[0];
  float* op = out + ((bat*N_SEQ + (q0 + wave*16 + ln))*N_HEAD + head)*HDIM + 8*hh;
#pragma unroll
  for(int db=0; db<4; db++){
    float4 lo, hi;
    lo.x=O[db][0]*inv; lo.y=O[db][1]*inv; lo.z=O[db][2]*inv; lo.w=O[db][3]*inv;
    hi.x=O[db][4]*inv; hi.y=O[db][5]*inv; hi.z=O[db][6]*inv; hi.w=O[db][7]*inv;
    *(float4*)(op + 16*db)     = lo;
    *(float4*)(op + 16*db + 4) = hi;
  }
}

extern "C" void kernel_launch(void* const* d_in, const int* in_sizes, int n_in,
                              void* d_out, int out_size, void* d_ws, size_t ws_size,
                              hipStream_t stream) {
  (void)in_sizes; (void)n_in; (void)out_size; (void)d_ws; (void)ws_size;
  const float* q = (const float*)d_in[0];
  const float* k = (const float*)d_in[1];
  const float* v = (const float*)d_in[2];
  float* out = (float*)d_out;
  // 3 groups x 2 batches x 4 heads x 64 query-blocks = 1536 workgroups
  RingDilatedAttention_8615704396434_kernel<<<dim3(1536), dim3(256), 0, stream>>>(q, k, v, out);
}
